// GPT2_11974368821412
// MI455X (gfx1250) — matrix operand order repaired and checked by reading, not compiled, not run on hardware
//
#include <hip/hip_runtime.h>
#include <math.h>

// GPT-2 forward for MI455X (gfx1250, wave32, WMMA bf16, async-LDS copies).
// Input order assumption (setup_inputs dict insertion order, flattened):
// 0 idx, 1 targets, 2 wte, 3 wpe, 4 w_lm, 5 ln_f_g, 6 ln_f_b,
// then 8 blocks x {wq,wk,wv,bq,bk,bv,wo,bo,w1,b1,w2,b2,ln1_g,ln1_b,ln2_g,ln2_b}.

#define Bsz   4
#define Tlen  1024
#define Dm    768
#define Hh    12
#define HDim  64
#define Ll    8
#define Vv    50257
#define VP    50304            // padded LM-head N (= 393*128)
#define NTOK  (Bsz*Tlen)       // 4096
#define FF    (4*Dm)           // 3072
#define EPSV  1e-5f
#define SCALEV 0.125f          // 1/sqrt(64)

typedef __attribute__((ext_vector_type(16))) __bf16 bf16x16;
typedef __attribute__((ext_vector_type(8)))  __bf16 bf16x8;
typedef __attribute__((ext_vector_type(8)))  float  f32x8;

__device__ __forceinline__ f32x8 wmma_bf16(bf16x16 a, bf16x16 b, f32x8 c){
  return __builtin_amdgcn_wmma_f32_16x16x32_bf16(false, a, false, b, (short)0, c, false, false);
}
__device__ __forceinline__ bf16x16 join8(bf16x8 lo, bf16x8 hi){
  union { bf16x16 v; bf16x8 h[2]; } u; u.h[0]=lo; u.h[1]=hi; return u.v;
}
__device__ __forceinline__ bf16x8 ld8(const __bf16* p){ return *(const bf16x8*)p; }
__device__ __forceinline__ f32x8 zero8(){
  f32x8 v;
  #pragma unroll
  for (int i=0;i<8;i++) v[i]=0.f;
  return v;
}

// --- async global->LDS copy (CDNA5 ASYNCcnt path) with portable fallback ---
#if defined(__has_builtin)
#  if __has_builtin(__builtin_amdgcn_global_load_async_to_lds_b128)
#    define ASYNC_CP 1
#  endif
#endif

typedef int v4i __attribute__((vector_size(16)));
typedef __attribute__((address_space(1))) v4i* gptr_v4i;
typedef __attribute__((address_space(3))) v4i* lptr_v4i;

__device__ __forceinline__ void async_cp_b128(const __bf16* g, __bf16* l){
#ifdef ASYNC_CP
  __builtin_amdgcn_global_load_async_to_lds_b128(
      (gptr_v4i)(unsigned long long)(const void*)g,
      (lptr_v4i)(unsigned int)(unsigned long long)(const void*)l,
      0, 0);
#else
  *(bf16x8*)l = *(const bf16x8*)g;
#endif
}
__device__ __forceinline__ void async_wait(){
#ifdef ASYNC_CP
  __asm__ volatile("s_wait_asynccnt 0x0" ::: "memory");
#endif
}

// ---------------- embeddings ----------------
__global__ __launch_bounds__(256) void k_embed(const int* __restrict__ idx,
                                               const float* __restrict__ wte,
                                               const float* __restrict__ wpe,
                                               float* __restrict__ x){
  int bt = blockIdx.x; int t = bt % Tlen;
  long tok = idx[bt];
  const float* we = wte + (size_t)tok*Dm;
  const float* pe = wpe + (size_t)t*Dm;
  float* xo = x + (size_t)bt*Dm;
  for (int i = threadIdx.x; i < Dm; i += 256) xo[i] = we[i] + pe[i];
}

// ---------------- layernorm (row per block) ----------------
__global__ __launch_bounds__(256) void k_ln(const float* __restrict__ x,
                                            const float* __restrict__ g,
                                            const float* __restrict__ b,
                                            float* __restrict__ y){
  __shared__ float s1[256], s2[256];
  int row = blockIdx.x;
  const float* xr = x + (size_t)row*Dm;
  float sum=0.f, sq=0.f;
  for (int i=threadIdx.x;i<Dm;i+=256){ float v=xr[i]; sum+=v; sq+=v*v; }
  s1[threadIdx.x]=sum; s2[threadIdx.x]=sq; __syncthreads();
  for (int off=128; off>0; off>>=1){
    if (threadIdx.x < off){ s1[threadIdx.x]+=s1[threadIdx.x+off]; s2[threadIdx.x]+=s2[threadIdx.x+off]; }
    __syncthreads();
  }
  float mean = s1[0]*(1.f/Dm);
  float var  = s2[0]*(1.f/Dm) - mean*mean;
  float r = rsqrtf(var + EPSV);
  float* yr = y + (size_t)row*Dm;
  for (int i=threadIdx.x;i<Dm;i+=256) yr[i] = (xr[i]-mean)*r*g[i] + b[i];
}

// ---------- weight conversion fp32 -> bf16, stored TRANSPOSED as WT[N][K] ----------
// wq/wk/wv: (H, D, HD) -> WT[n=h*64+k][d]
__global__ __launch_bounds__(256) void k_w_qkvT(const float* __restrict__ in,
                                                __bf16* __restrict__ out){
  int i = blockIdx.x*256 + threadIdx.x;
  if (i >= Dm*Dm) return;
  int n = i / Dm, d = i % Dm; int h = n >> 6, kk = n & 63;
  out[i] = (__bf16)in[((size_t)h*Dm + d)*HDim + kk];
}
// generic [K][N] -> WT[N][K]
__global__ __launch_bounds__(256) void k_w_T(const float* __restrict__ in,
                                             __bf16* __restrict__ out, int K, int N){
  long i = (long)blockIdx.x*256 + threadIdx.x;
  if (i >= (long)K*N) return;
  int n = (int)(i / K), k = (int)(i % K);
  out[i] = (__bf16)in[(size_t)k*N + n];
}
// lm head [D][V] -> WT[VP][D], zero rows for n >= V
__global__ __launch_bounds__(256) void k_w_lmT(const float* __restrict__ in,
                                               __bf16* __restrict__ out){
  long i = (long)blockIdx.x*256 + threadIdx.x;
  long tot = (long)VP*Dm; if (i >= tot) return;
  int n = (int)(i / Dm), k = (int)(i % Dm);
  out[i] = (n < Vv) ? (__bf16)in[(size_t)k*Vv + n] : (__bf16)0.f;
}

// q,k,v fp32 [B,T,H*HD] -> bf16 Q/K [bh][T][HD], V transposed [bh][HD][T]
__global__ __launch_bounds__(256) void k_qkv_cvt(const float* __restrict__ qf,
                                                 const float* __restrict__ kf,
                                                 const float* __restrict__ vf,
                                                 __bf16* __restrict__ Qb,
                                                 __bf16* __restrict__ Kb,
                                                 __bf16* __restrict__ VbT){
  int i = blockIdx.x*256 + threadIdx.x;
  if (i >= NTOK*Dm) return;
  int bt = i / Dm, c = i % Dm;
  int b = bt / Tlen, t = bt % Tlen;
  int h = c >> 6, k = c & 63;
  int bh = b*Hh + h;
  Qb [((size_t)bh*Tlen + t)*HDim + k] = (__bf16)qf[i];
  Kb [((size_t)bh*Tlen + t)*HDim + k] = (__bf16)kf[i];
  VbT[((size_t)bh*HDim + k)*Tlen + t] = (__bf16)vf[i];
}

// ------- WMMA bf16 GEMM: C = act(A[M,K] * W[K,N] + bias (+res)), W given as WT[N][K] -------
// Block tile 128x128x32, 8 waves, each wave 32x64 (8 wmma per K-step).
#define BM 128
#define BN 128
#define LDSA 48
#define LDSB 48
__global__ __launch_bounds__(256) void k_gemm(
    const float*  __restrict__ A, int lda,
    const __bf16* __restrict__ WT, int ldb,   // ldb = K
    const float*  __restrict__ bias,
    const float*  __restrict__ res,
    float*        __restrict__ C, int ldc,
    int M, int N, int K, int relu)
{
  __shared__ __attribute__((aligned(16))) __bf16 As [BM*LDSA];
  __shared__ __attribute__((aligned(16))) __bf16 BsT[BN*LDSB];
  int tid = threadIdx.x;
  int n0 = blockIdx.x*BN, m0 = blockIdx.y*BM;
  int lane = tid & 31, w = tid >> 5;
  int wm = (w & 3)*32, wn = (w >> 2)*64;
  int l15 = lane & 15; int hi = (lane >> 4) & 1;
  f32x8 acc[2][4];
  #pragma unroll
  for (int am=0;am<2;am++)
    #pragma unroll
    for (int nt=0;nt<4;nt++) acc[am][nt] = zero8();

  int ar = tid >> 1, ac  = (tid & 1)*16;   // A tile: 128 rows x 32 cols, 16 f32/thread
  int bn = tid >> 1, bkc = (tid & 1)*16;   // B tile: 128 rows(N) x 32 cols(K), 16 bf16/thread

  for (int k0 = 0; k0 < K; k0 += 32){
    { // B tile: contiguous 16B chunks -> LDS (async DMA when available)
      const __bf16* bp = WT + (size_t)(n0+bn)*ldb + k0 + bkc;
      __bf16* lp = BsT + bn*LDSB + bkc;
      async_cp_b128(bp,   lp);
      async_cp_b128(bp+8, lp+8);
      if (k0 + 32 < K) __builtin_prefetch(bp + 32, 0, 1);
    }
    { // A tile: fp32 -> bf16 into LDS (M always multiple of 128: no guard)
      const float* ap = A + (size_t)(m0+ar)*lda + k0 + ac;
      __bf16* dst = As + ar*LDSA + ac;
      #pragma unroll
      for (int i=0;i<16;i++) dst[i] = (__bf16)ap[i];
      if (k0 + 32 < K) __builtin_prefetch(ap + 32, 0, 1);
    }
    async_wait();
    __syncthreads();
    // A fragments: row = wm+am*16+l15; lane<16 -> K 0..7,16..23 ; lane>=16 -> K 8..15,24..31
    bf16x16 af[2];
    #pragma unroll
    for (int am=0;am<2;am++){
      const __bf16* arow = As + (wm + am*16 + l15)*LDSA;
      af[am] = join8(ld8(arow + (hi?8:0)), ld8(arow + (hi?24:16)));
    }
    // B fragments: col = wn+nt*16+l15; lane<16 -> K 0..7,16..23 ; lane>=16 -> K 8..15,24..31
    #pragma unroll
    for (int nt=0;nt<4;nt++){
      const __bf16* bp = BsT + (wn + nt*16 + l15)*LDSB;
      bf16x16 bfr = join8(ld8(bp + (hi?8:0)), ld8(bp + (hi?24:16)));
      acc[0][nt] = wmma_bf16(af[0], bfr, acc[0][nt]);
      acc[1][nt] = wmma_bf16(af[1], bfr, acc[1][nt]);
    }
    __syncthreads();
  }
  #pragma unroll
  for (int am=0;am<2;am++){
    #pragma unroll
    for (int r=0;r<8;r++){
      int m = m0 + wm + am*16 + r + (hi?8:0);
      #pragma unroll
      for (int nt=0;nt<4;nt++){
        int n = n0 + wn + nt*16 + l15;
        if (n >= N) continue;
        float v = acc[am][nt][r];
        if (bias) v += bias[n];
        if (res)  v += res[(size_t)m*ldc + n];
        if (relu) v = fmaxf(v, 0.f);
        C[(size_t)m*ldc + n] = v;
      }
    }
  }
  (void)M;
}

// ---------------- flash attention (one 16-query tile per wave) ----------------
__global__ __launch_bounds__(256) void k_attn(const __bf16* __restrict__ Qb,
                                              const __bf16* __restrict__ Kb,
                                              const __bf16* __restrict__ VbT,
                                              float* __restrict__ O){
  __shared__ __attribute__((aligned(16))) __bf16 Pst[8][16*32]; // per-wave P staging
  int wid = threadIdx.x >> 5, lane = threadIdx.x & 31;
  int g = blockIdx.x*8 + wid;       // global q-tile id
  int bh = g >> 6;                  // (b*H + h), T/16 = 64 tiles each
  int qt = g & 63;
  int qbase = qt * 16;
  int b = bh / Hh, h = bh % Hh;
  int l15 = lane & 15; int hi = (lane >> 4) & 1;
  const __bf16* Q  = Qb  + (size_t)bh*Tlen*HDim;
  const __bf16* Kp = Kb  + (size_t)bh*Tlen*HDim;
  const __bf16* Vt = VbT + (size_t)bh*HDim*Tlen;
  __bf16* P = Pst[wid];

  const __bf16* qrow = Q + (size_t)(qbase + l15)*HDim;
  bf16x16 qfrag[2];
  #pragma unroll
  for (int j=0;j<2;j++)
    qfrag[j] = join8(ld8(qrow + j*32 + (hi?8:0)), ld8(qrow + j*32 + (hi?24:16)));

  f32x8 oacc[4];
  #pragma unroll
  for (int nt=0;nt<4;nt++) oacc[nt] = zero8();
  float mrow[8], lrow[8];
  #pragma unroll
  for (int r=0;r<8;r++){ mrow[r] = -INFINITY; lrow[r] = 0.f; }

  int kend = qbase + 16;            // causal key limit (exclusive)
  for (int kt = 0; kt < kend; kt += 32){
    f32x8 s0 = zero8(), s1 = zero8();
    #pragma unroll
    for (int j=0;j<2;j++){
      const __bf16* kp0 = Kp + (size_t)(kt      + l15)*HDim + j*32;
      s0 = wmma_bf16(qfrag[j], join8(ld8(kp0 + (hi?8:0)), ld8(kp0 + (hi?24:16))), s0);
      const __bf16* kp1 = Kp + (size_t)(kt + 16 + l15)*HDim + j*32;
      s1 = wmma_bf16(qfrag[j], join8(ld8(kp1 + (hi?8:0)), ld8(kp1 + (hi?24:16))), s1);
    }
    float alpha[8];
    #pragma unroll
    for (int r=0;r<8;r++){
      int q = qbase + r + (hi?8:0);
      float v0 = s0[r]*SCALEV; if (kt      + l15 > q) v0 = -INFINITY;
      float v1 = s1[r]*SCALEV; if (kt + 16 + l15 > q) v1 = -INFINITY;
      float mx = fmaxf(v0, v1);
      #pragma unroll
      for (int off=1; off<16; off<<=1) mx = fmaxf(mx, __shfl_xor(mx, off, 32));
      float mnew = fmaxf(mrow[r], mx);
      float a  = __expf(mrow[r] - mnew);
      float p0 = __expf(v0 - mnew);
      float p1 = __expf(v1 - mnew);
      float ps = p0 + p1;
      #pragma unroll
      for (int off=1; off<16; off<<=1) ps += __shfl_xor(ps, off, 32);
      lrow[r] = lrow[r]*a + ps;
      mrow[r] = mnew;
      alpha[r] = a;
      s0[r] = p0; s1[r] = p1;
    }
    #pragma unroll
    for (int nt=0;nt<4;nt++){
      #pragma unroll
      for (int r=0;r<8;r++) oacc[nt][r] *= alpha[r];
    }
    #pragma unroll
    for (int r=0;r<8;r++){
      int prow = r + (hi?8:0);
      P[prow*32 +      l15] = (__bf16)s0[r];
      P[prow*32 + 16 + l15] = (__bf16)s1[r];
    }
    __asm__ volatile("" ::: "memory");
    const __bf16* pr = P + l15*32;
    bf16x16 pfrag = join8(ld8(pr + (hi?8:0)), ld8(pr + (hi?24:16)));
    __asm__ volatile("" ::: "memory");
    #pragma unroll
    for (int nt=0;nt<4;nt++){
      const __bf16* vp = Vt + (size_t)(nt*16 + l15)*Tlen + kt;
      oacc[nt] = wmma_bf16(pfrag, join8(ld8(vp + (hi?8:0)), ld8(vp + (hi?24:16))), oacc[nt]);
    }
  }
  #pragma unroll
  for (int r=0;r<8;r++){
    int t = qbase + r + (hi?8:0);
    float inv = 1.f / lrow[r];
    float* orow = O + ((size_t)b*Tlen + t)*Dm + h*HDim;
    #pragma unroll
    for (int nt=0;nt<4;nt++) orow[nt*16 + l15] = oacc[nt][r]*inv;
  }
}

// ---------------- NLL ----------------
__global__ __launch_bounds__(256) void k_nll(const float* __restrict__ logits,
                                             const int* __restrict__ tgt,
                                             float* __restrict__ nll){
  __shared__ float sh[256];
  int row = blockIdx.x;
  const float* lr = logits + (size_t)row*Vv;
  float mx = -INFINITY;
  for (int i=threadIdx.x;i<Vv;i+=256) mx = fmaxf(mx, lr[i]);
  sh[threadIdx.x]=mx; __syncthreads();
  for (int off=128;off>0;off>>=1){
    if (threadIdx.x<off) sh[threadIdx.x]=fmaxf(sh[threadIdx.x],sh[threadIdx.x+off]);
    __syncthreads();
  }
  mx = sh[0]; __syncthreads();
  float sum=0.f;
  for (int i=threadIdx.x;i<Vv;i+=256) sum += __expf(lr[i]-mx);
  sh[threadIdx.x]=sum; __syncthreads();
  for (int off=128;off>0;off>>=1){
    if (threadIdx.x<off) sh[threadIdx.x]+=sh[threadIdx.x+off];
    __syncthreads();
  }
  if (threadIdx.x==0){
    float lse = __logf(sh[0]) + mx;
    nll[row] = lse - lr[tgt[row]];
  }
}

__global__ __launch_bounds__(256) void k_loss(const float* __restrict__ nll,
                                              float* __restrict__ out){
  __shared__ float sh[256];
  float s=0.f;
  for (int i=threadIdx.x;i<NTOK;i+=256) s += nll[i];
  sh[threadIdx.x]=s; __syncthreads();
  for (int off=128;off>0;off>>=1){
    if (threadIdx.x<off) sh[threadIdx.x]+=sh[threadIdx.x+off];
    __syncthreads();
  }
  if (threadIdx.x==0) out[0] = sh[0] * (1.f/(float)NTOK);
}

// ---------------- host ----------------
extern "C" void kernel_launch(void* const* d_in, const int* in_sizes, int n_in,
                              void* d_out, int out_size, void* d_ws, size_t ws_size,
                              hipStream_t stream)
{
  (void)in_sizes; (void)n_in; (void)out_size; (void)ws_size;
  const int*   idx  = (const int*)d_in[0];
  const int*   tgt  = (const int*)d_in[1];
  const float* wte  = (const float*)d_in[2];
  const float* wpe  = (const float*)d_in[3];
  const float* wlm  = (const float*)d_in[4];
  const float* lnfg = (const float*)d_in[5];
  const float* lnfb = (const float*)d_in[6];
  auto blkp = [&](int l,int j){ return (const float*)d_in[7 + l*16 + j]; };

  char* p = (char*)d_ws;
  auto carve = [&](size_t bytes)->void*{
    void* r = (void*)p; p += (bytes + 255) & ~(size_t)255; return r;
  };
  float* x    = (float*)carve((size_t)NTOK*Dm*4);
  float* hbuf = (float*)carve((size_t)NTOK*Dm*4);
  float* qf   = (float*)carve((size_t)NTOK*Dm*4);
  float* kf   = (float*)carve((size_t)NTOK*Dm*4);
  float* vf   = (float*)carve((size_t)NTOK*Dm*4);
  float* oat  = (float*)carve((size_t)NTOK*Dm*4);
  float* hid  = (float*)carve((size_t)NTOK*FF*4);
  float* nll  = (float*)carve((size_t)NTOK*4);
  __bf16* Qbf = (__bf16*)carve((size_t)Bsz*Hh*Tlen*HDim*2);
  __bf16* Kbf = (__bf16*)carve((size_t)Bsz*Hh*Tlen*HDim*2);
  __bf16* VbT = (__bf16*)carve((size_t)Bsz*Hh*HDim*Tlen*2);
  __bf16* Wlmb= (__bf16*)carve((size_t)VP*Dm*2);
  __bf16 *Wq[Ll], *Wk[Ll], *Wv[Ll], *Wo[Ll], *W1[Ll], *W2[Ll];
  for (int l=0;l<Ll;l++){
    Wq[l]=(__bf16*)carve((size_t)Dm*Dm*2);
    Wk[l]=(__bf16*)carve((size_t)Dm*Dm*2);
    Wv[l]=(__bf16*)carve((size_t)Dm*Dm*2);
    Wo[l]=(__bf16*)carve((size_t)Dm*Dm*2);
    W1[l]=(__bf16*)carve((size_t)Dm*FF*2);
    W2[l]=(__bf16*)carve((size_t)FF*Dm*2);
  }
  float* logits = (float*)d_out;
  float* lossp  = logits + (size_t)NTOK*Vv;

  // --- weights -> bf16, transposed to WT[N][K] (once; ~182MB bf16 ~ fits global L2) ---
  {
    int n  = Dm*Dm;  int g  = (n  + 255)/256;
    int n2 = Dm*FF;  int g2 = (n2 + 255)/256;
    for (int l=0;l<Ll;l++){
      k_w_qkvT<<<g ,256,0,stream>>>(blkp(l,0),  Wq[l]);
      k_w_qkvT<<<g ,256,0,stream>>>(blkp(l,1),  Wk[l]);
      k_w_qkvT<<<g ,256,0,stream>>>(blkp(l,2),  Wv[l]);
      k_w_T  <<<g ,256,0,stream>>>(blkp(l,6),  Wo[l], Dm, Dm);
      k_w_T  <<<g2,256,0,stream>>>(blkp(l,8),  W1[l], Dm, FF);
      k_w_T  <<<g2,256,0,stream>>>(blkp(l,10), W2[l], FF, Dm);
    }
    long nl = (long)VP*Dm; int gl = (int)((nl + 255)/256);
    k_w_lmT<<<gl,256,0,stream>>>(wlm, Wlmb);
  }

  k_embed<<<NTOK,256,0,stream>>>(idx, wte, wpe, x);

  auto gemm = [&](const float* A,int lda,const __bf16* WT,const float* bias,
                  const float* res,float* C,int ldc,int M,int N,int K,int relu){
    dim3 grid((N+BN-1)/BN,(M+BM-1)/BM);
    k_gemm<<<grid,256,0,stream>>>(A,lda,WT,K,bias,res,C,ldc,M,N,K,relu);
  };

  for (int l=0;l<Ll;l++){
    k_ln<<<NTOK,256,0,stream>>>(x, blkp(l,12), blkp(l,13), hbuf);
    gemm(hbuf,Dm, Wq[l], blkp(l,3), nullptr, qf, Dm, NTOK,Dm,Dm, 0);
    gemm(hbuf,Dm, Wk[l], blkp(l,4), nullptr, kf, Dm, NTOK,Dm,Dm, 0);
    gemm(hbuf,Dm, Wv[l], blkp(l,5), nullptr, vf, Dm, NTOK,Dm,Dm, 0);
    { int n=NTOK*Dm; k_qkv_cvt<<<(n+255)/256,256,0,stream>>>(qf,kf,vf,Qbf,Kbf,VbT); }
    k_attn<<<(Bsz*Hh*(Tlen/16))/8,256,0,stream>>>(Qbf,Kbf,VbT,oat);
    gemm(oat ,Dm, Wo[l], blkp(l,7),  x, x, Dm, NTOK,Dm,Dm, 0);           // +residual
    k_ln<<<NTOK,256,0,stream>>>(x, blkp(l,14), blkp(l,15), hbuf);
    gemm(hbuf,Dm, W1[l], blkp(l,9),  nullptr, hid, FF, NTOK,FF,Dm, 1);   // relu
    gemm(hid ,FF, W2[l], blkp(l,11), x, x, Dm, NTOK,Dm,FF, 0);           // +residual
  }
  k_ln<<<NTOK,256,0,stream>>>(x, lnfg, lnfb, hbuf);
  gemm(hbuf,Dm, Wlmb, nullptr, nullptr, logits, Vv, NTOK,Vv,Dm, 0);
  k_nll<<<NTOK,256,0,stream>>>(logits, tgt, nll);
  k_loss<<<1,256,0,stream>>>(nll, lossp);
}
